// MultiHeadAttention_47966194761822
// MI455X (gfx1250) — matrix operand order repaired and checked by reading, not compiled, not run on hardware
//
#include <hip/hip_runtime.h>

// ---------------------------------------------------------------------------
// MultiHeadAttention on gfx1250 (MI455X): all matmuls via v_wmma_f32_16x16x32_bf16
// B=2, N=2048, E=1024, H=16, D=64, SCALE = 1024^-0.5 = 1/32
// Compute-bound (~69 GFLOP vs ~60MB HBM traffic) -> bf16 WMMA, fp32 accumulate.
// K/V tiles stream through double-buffered LDS via the async copy engine
// (ASYNCcnt); softmax row-sums are folded into the PV WMMA via a ones-column.
// Ping-pong selection is done on INTEGER offsets into single __shared__ arrays
// (pointer-level selects would break address-space inference -> flat loads).
// ---------------------------------------------------------------------------

#define B_SZ   2
#define N_SEQ  2048
#define EMB    1024
#define NHEAD  16
#define HDIM   64
#define SCALEF 0.03125f

typedef __attribute__((ext_vector_type(16))) __bf16          v16bf;
typedef __attribute__((ext_vector_type(16))) unsigned short  v16u;
typedef __attribute__((ext_vector_type(8)))  float           v8f;
typedef __attribute__((ext_vector_type(4)))  unsigned int    u32x4;
typedef __attribute__((ext_vector_type(4)))  float           f32x4;
typedef int gvec4i __attribute__((__vector_size__(16)));   // async builtin pointee

union V16 { v16u u; v16bf bf; u32x4 q[2]; };

#define AS1 __attribute__((address_space(1)))
#define AS3 __attribute__((address_space(3)))

#ifdef __has_builtin
#if __has_builtin(__builtin_amdgcn_global_load_async_to_lds_b128) && \
    __has_builtin(__builtin_amdgcn_s_wait_asynccnt)
#define USE_ASYNC 1
#endif
#endif
#ifndef USE_ASYNC
#define USE_ASYNC 0
#endif

// 16-byte global->LDS copy (async DMA when available)
__device__ __forceinline__ void cp16(unsigned short* dst, const unsigned short* src) {
#if USE_ASYNC
  __builtin_amdgcn_global_load_async_to_lds_b128(
      (AS1 gvec4i*)src, (AS3 gvec4i*)dst, 0, 0);
#else
  *(u32x4*)dst = *(const u32x4*)src;
#endif
}
__device__ __forceinline__ void wait_async() {
#if USE_ASYNC
  __builtin_amdgcn_s_wait_asynccnt(0);
#endif
}

// fp32 -> bf16 (round-to-nearest-even), as ushort bit pattern
__device__ __forceinline__ unsigned short f2bf(float f) {
  unsigned int x = __float_as_uint(f);
  x += 0x7FFFu + ((x >> 16) & 1u);
  return (unsigned short)(x >> 16);
}
__device__ __forceinline__ unsigned short to_bf(float x)          { return f2bf(x); }
__device__ __forceinline__ unsigned short to_bf(unsigned short x) { return x; }

__device__ __forceinline__ unsigned int pack2(unsigned short a, unsigned short b) {
  return (unsigned int)a | ((unsigned int)b << 16);
}
__device__ __forceinline__ u32x4 load8bf(const float* p) {
  f32x4 f0 = *(const f32x4*)p;
  f32x4 f1 = *(const f32x4*)(p + 4);
  u32x4 r;
  r.x = pack2(f2bf(f0.x), f2bf(f0.y));
  r.y = pack2(f2bf(f0.z), f2bf(f0.w));
  r.z = pack2(f2bf(f1.x), f2bf(f1.y));
  r.w = pack2(f2bf(f1.z), f2bf(f1.w));
  return r;
}
__device__ __forceinline__ u32x4 load8bf(const unsigned short* p) {
  return *(const u32x4*)p;
}

// ---------------------------------------------------------------------------
// Projection GEMM:  C[M x NOUT] = A[M x KDIM] * W[KDIM x NOUT] + bias
// MODE 0: Q proj  -> bf16 xq[b][h][n][d], pre-scaled by SCALEF
// MODE 1: KV proj -> bf16 xk[b][h][n][d] (cols < 1024), xv[b][h][d][n] (transposed)
// MODE 2: out proj (A = bf16 attn) -> fp32 row-major oF[M x NOUT]
// 128 threads (4 waves), 64x64 tile, K stepped by 32, ping-pong LDS staging
// (one barrier per K-iter; staging of tile i+1 overlaps compute of tile i).
// ---------------------------------------------------------------------------
template <typename TA, int MODE, int KDIM, int NOUT>
__global__ __launch_bounds__(128)
void gemm_proj(const TA* __restrict__ A, const float* __restrict__ W,
               const float* __restrict__ bias,
               unsigned short* __restrict__ oQ,
               unsigned short* __restrict__ oK,
               unsigned short* __restrict__ oV,
               float* __restrict__ oF)
{
  // [buf][64 rows][pitch 40] and fragment-ordered [buf][4*32 lanes][16 halves]
  __shared__ __attribute__((aligned(16))) unsigned short AsAll[2 * 2560];
  __shared__ __attribute__((aligned(16))) unsigned short WfAll[2 * 2048];

  const int t    = threadIdx.x;
  const int lane = t & 31;
  const int w    = t >> 5;
  const int g    = lane >> 4;
  const int lc   = lane & 15;
  const int m0   = blockIdx.y * 64;
  const int c0   = blockIdx.x * 64;

  // ---- hoisted staging pointers/offsets -----------------------------------
  // A tile: thread owns rows (t>>2) and 32+(t>>2), 8-elem segment (t&3)
  const TA* aS = A + (size_t)(m0 + (t >> 2)) * KDIM + (t & 3) * 8;
  const int aOff = (t >> 2) * 40 + (t & 3) * 8;
  // W tile: thread owns column c = t&63, rows k = (t>>6) + 2i
  const float* wS = W + (size_t)(t >> 6) * NOUT + c0 + (t & 63);
  const int wOff = ((((t & 63) >> 4) * 32) + (t & 15)) * 16 + (t >> 6);

  auto stage = [&](int buf) {
    const int ab = buf * 2560 + aOff;
    if (sizeof(TA) == 2) {
      cp16(&AsAll[ab],        (const unsigned short*)aS);
      cp16(&AsAll[ab + 1280], (const unsigned short*)(aS + (size_t)32 * KDIM));
    } else {
      *(u32x4*)&AsAll[ab]        = load8bf(aS);
      *(u32x4*)&AsAll[ab + 1280] = load8bf(aS + (size_t)32 * KDIM);
    }
    aS += 32;
    const int wb = buf * 2048 + wOff;
#pragma unroll
    for (int i = 0; i < 16; ++i)   // dst & src offsets are compile-time consts
      WfAll[wb + 256 * (i >> 3) + 2 * (i & 7)] = to_bf(wS[(size_t)(2 * i) * NOUT]);
    __builtin_prefetch(wS + (size_t)32 * NOUT, 0, 1);  // speculative, next tile
    wS += (size_t)32 * NOUT;
  };

  // ---- hoisted fragment base offsets --------------------------------------
  const int aF = (w * 16 + lc) * 40 + 16 * g;
  const int bF = lane * 16;

  v8f acc[4] = {};
  constexpr int iters = KDIM >> 5;

  stage(0);
  for (int it = 0; it < iters; ++it) {
    const int cur = it & 1;
    if (sizeof(TA) == 2) wait_async();
    __syncthreads();                         // staged tile visible to all waves
    if (it + 1 < iters) stage(1 - cur);

    const int ca = cur * 2560 + aF;
    const int cb = cur * 2048 + bF;
    V16 af;
    af.q[0] = *(const u32x4*)&AsAll[ca];
    af.q[1] = *(const u32x4*)&AsAll[ca + 8];
#pragma unroll
    for (int j = 0; j < 4; ++j) {
      V16 bm;
      bm.q[0] = *(const u32x4*)&WfAll[cb + j * 512];
      bm.q[1] = *(const u32x4*)&WfAll[cb + j * 512 + 8];
      acc[j] = __builtin_amdgcn_wmma_f32_16x16x32_bf16(
          false, af.bf, false, bm.bf, (short)0, acc[j], false, false);
    }
  }

  // ---- epilogue: bias + scatter ------------------------------------------
#pragma unroll
  for (int j = 0; j < 4; ++j) {
    const int c  = c0 + j * 16 + lc;
    const float bv = bias[c];
#pragma unroll
    for (int r = 0; r < 8; ++r) {
      const int   m   = m0 + w * 16 + r + 8 * g;       // C layout row = r + 8g
      const float val = acc[j][r] + bv;
      if (MODE == 0) {
        const int hd = c >> 6, d = c & 63;
        const int bb = m >> 11, n = m & (N_SEQ - 1);
        oQ[(((size_t)(bb * NHEAD + hd)) * N_SEQ + n) * HDIM + d] = f2bf(val * SCALEF);
      } else if (MODE == 1) {
        const int bb = m >> 11, n = m & (N_SEQ - 1);
        if (c < EMB) {
          const int hd = c >> 6, d = c & 63;
          oK[(((size_t)(bb * NHEAD + hd)) * N_SEQ + n) * HDIM + d] = f2bf(val);
        } else {
          const int c2 = c - EMB, hd = c2 >> 6, d = c2 & 63;
          oV[(((size_t)(bb * NHEAD + hd)) * HDIM + d) * N_SEQ + n] = f2bf(val);
        }
      } else {
        oF[(size_t)m * NOUT + c] = val;
      }
    }
  }
}

// ---------------------------------------------------------------------------
// Flash attention, "+1 softmax", max clipped at 0:
//   denom = sum exp(x - m) + exp(-m),  m = max(rowmax, 0)
// m_init = 0; the running row-sum is accumulated BY THE MATRIX ENGINE through a
// constant ones-column V tile (acc tile j=4), under the same rescaling chain as
// the output; exp(-m_final) is added once at the end.  Only the row-max needs
// cross-lane shuffles.  K/V tiles: double-buffered async LDS staging, one
// block barrier per 32-key tile; all tile offsets are immediates.
// ---------------------------------------------------------------------------
__global__ __launch_bounds__(128)
void attn_kernel(const unsigned short* __restrict__ xq,   // [b][h][n][d]
                 const unsigned short* __restrict__ xk,   // [b][h][n][d]
                 const unsigned short* __restrict__ xvT,  // [b][h][d][n]
                 const int* __restrict__ mask,            // 0/1 per (b, key)
                 unsigned short* __restrict__ attn)       // bf16 [b][n][h*64+d]
{
  // [buf][32 keys][pitch 72] / [buf][64 d][pitch 40]
  __shared__ __attribute__((aligned(16))) unsigned short KtAll[2 * 2304];
  __shared__ __attribute__((aligned(16))) unsigned short VtAll[2 * 2560];
  __shared__ __attribute__((aligned(16))) unsigned short Vones[16 * 40];  // d-row 0 = 1.0
  __shared__ __attribute__((aligned(16))) unsigned short Pt[4][16 * 40];

  const int t    = threadIdx.x;
  const int lane = t & 31;
  const int w    = t >> 5;
  const int g    = lane >> 4;
  const int lc   = lane & 15;
  const int bh   = blockIdx.y;
  const int b    = bh >> 4, h = bh & 15;
  const int q0   = blockIdx.x * 64;
  const size_t headoff = (size_t)bh * N_SEQ * HDIM;

  // ones-column tile for the row-sum accumulator (read-only after init)
  for (int i = t; i < 16 * 40; i += 128)
    Vones[i] = (i < 32) ? (unsigned short)0x3F80 : (unsigned short)0;

  // ---- hoisted staging pointers/offsets -----------------------------------
  const unsigned short* srcK = xk + headoff + (size_t)(t >> 3) * HDIM + (t & 7) * 8;
  const unsigned short* srcV = xvT + (size_t)bh * HDIM * N_SEQ
                                   + (size_t)(t >> 2) * N_SEQ + (t & 3) * 8;
  const int tK = (t >> 3) * 72 + (t & 7) * 8;
  const int tV = (t >> 2) * 40 + (t & 3) * 8;

  auto stage = [&](int buf) {
    const int kb_ = buf * 2304 + tK;
    const int vb_ = buf * 2560 + tV;
    cp16(&KtAll[kb_],        srcK);
    cp16(&KtAll[kb_ + 1152], srcK + 16 * HDIM);
    cp16(&VtAll[vb_],        srcV);
    cp16(&VtAll[vb_ + 1280], srcV + (size_t)32 * N_SEQ);
    srcK += 32 * HDIM;   // next 32 keys
    srcV += 32;          // next 32 key-columns
  };

  // ---- hoisted fragment base offsets --------------------------------------
  const int fK = lc * 72 + 16 * g;
  const int fV = lc * 40 + 16 * g;
  unsigned short* pw  = &Pt[w][0];
  unsigned short* pS  = pw + lc + g * 320;         // P store base
  const unsigned short* pA = pw + lc * 40 + 16 * g; // P A-frag base
  const int* mp = mask + (size_t)b * N_SEQ + lc;

  // Q fragments (two 16x32 A-frags covering d=0..63): 2x b128 each, from global
  V16 aq[2];
  {
    const unsigned short* qb = xq + headoff + (size_t)(q0 + w * 16 + lc) * HDIM;
#pragma unroll
    for (int c = 0; c < 2; ++c) {
      aq[c].q[0] = *(const u32x4*)(qb + c * 32 + 16 * g);
      aq[c].q[1] = *(const u32x4*)(qb + c * 32 + 8 + 16 * g);
    }
  }

  v8f acc[5] = {};             // j=0..3: 64 output cols; j=4: row-sum column
  float mrow[8];
#pragma unroll
  for (int r = 0; r < 8; ++r) mrow[r] = 0.0f;   // max clipped at 0

  constexpr int iters = N_SEQ / 32;

  stage(0);
  for (int it = 0; it < iters; ++it) {
    const int cur = it & 1;
    wait_async();
    __syncthreads();                          // current tile visible block-wide
    if (it + 1 < iters) stage(1 - cur);       // DMA overlaps compute

    const int kc = cur * 2304 + fK;
    const int vc = cur * 2560 + fV;

    // ---- S = Q K^T: two 16-key subtiles, K-dim 64 in two 32-chunks --------
    v8f st[2];
#pragma unroll
    for (int s = 0; s < 2; ++s) {
      v8f sa = {};
#pragma unroll
      for (int c = 0; c < 2; ++c) {
        V16 bk;
        bk.q[0] = *(const u32x4*)&KtAll[kc + s * 1152 + c * 32];
        bk.q[1] = *(const u32x4*)&KtAll[kc + s * 1152 + c * 32 + 8];
        sa = __builtin_amdgcn_wmma_f32_16x16x32_bf16(
            false, aq[c].bf, false, bk.bf, (short)0, sa, false, false);
      }
      st[s] = sa;
    }

    // ---- key mask: log(0) = -inf -> force column very negative ------------
#pragma unroll
    for (int s = 0; s < 2; ++s) {
      if (mp[s * 16] == 0) {
#pragma unroll
        for (int r = 0; r < 8; ++r) st[s][r] = -1e30f;
      }
    }
    mp += 32;

    // ---- online softmax: max-only cross-lane reduction --------------------
#pragma unroll
    for (int r = 0; r < 8; ++r) {
      float mx = fmaxf(st[0][r], st[1][r]);
#pragma unroll
      for (int off = 1; off < 16; off <<= 1)
        mx = fmaxf(mx, __shfl_xor(mx, off, 32));
      const float mn = fmaxf(mrow[r], mx);
      const float sc = __expf(mrow[r] - mn);
      mrow[r] = mn;
      const float p0 = __expf(st[0][r] - mn);
      const float p1 = __expf(st[1][r] - mn);
#pragma unroll
      for (int j = 0; j < 5; ++j) acc[j][r] *= sc;
      pS[r * 40]      = f2bf(p0);   // P row-major 16x32, const offsets
      pS[r * 40 + 16] = f2bf(p1);
    }
    __builtin_amdgcn_wave_barrier();  // order P stores before P reload (same wave)

    // ---- acc += P (16x32) * [V | ones] (32x(64+16)) ------------------------
    V16 ap;
    ap.q[0] = *(const u32x4*)(pA);
    ap.q[1] = *(const u32x4*)(pA + 8);
#pragma unroll
    for (int j = 0; j < 5; ++j) {
      V16 bv;
      if (j < 4) {
        bv.q[0] = *(const u32x4*)&VtAll[vc + j * 640];
        bv.q[1] = *(const u32x4*)&VtAll[vc + j * 640 + 8];
      } else {
        bv.q[0] = *(const u32x4*)&Vones[fV];
        bv.q[1] = *(const u32x4*)&Vones[fV + 8];
      }
      acc[j] = __builtin_amdgcn_wmma_f32_16x16x32_bf16(
          false, ap.bf, false, bv.bf, (short)0, acc[j], false, false);
    }
  }

  // ---- finalize: l = rowsum (acc[4] col 0 of each lane group) + exp(-m) ----
  float inv[8];
#pragma unroll
  for (int r = 0; r < 8; ++r) {
    const float rowsum = __shfl(acc[4][r], lane & 16, 32);  // broadcast from col 0
    inv[r] = 1.0f / (rowsum + __expf(-mrow[r]));
  }

  // ---- normalize, scatter to [b][n][h*64+d] bf16 ---------------------------
#pragma unroll
  for (int j = 0; j < 4; ++j)
#pragma unroll
    for (int r = 0; r < 8; ++r) {
      const int n = q0 + w * 16 + r + 8 * g;
      const int d = j * 16 + lc;
      attn[((size_t)b * N_SEQ + n) * EMB + h * HDIM + d] = f2bf(acc[j][r] * inv[r]);
    }
}

// ---------------------------------------------------------------------------
// Launcher. Inputs: q, kv, mask, Wq, bq, Wkv, bkv, Wo, bo
// ---------------------------------------------------------------------------
extern "C" void kernel_launch(void* const* d_in, const int* in_sizes, int n_in,
                              void* d_out, int out_size, void* d_ws, size_t ws_size,
                              hipStream_t stream) {
  (void)in_sizes; (void)n_in; (void)out_size; (void)ws_size;
  const float* q    = (const float*)d_in[0];
  const float* kv   = (const float*)d_in[1];
  const int*   mask = (const int*)  d_in[2];
  const float* Wq   = (const float*)d_in[3];
  const float* bq   = (const float*)d_in[4];
  const float* Wkv  = (const float*)d_in[5];
  const float* bkv  = (const float*)d_in[6];
  const float* Wo   = (const float*)d_in[7];
  const float* bo   = (const float*)d_in[8];
  float* out = (float*)d_out;

  const size_t headElems = (size_t)B_SZ * NHEAD * N_SEQ * HDIM;  // 4 Mi elems
  unsigned short* xq   = (unsigned short*)d_ws;
  unsigned short* xk   = xq + headElems;
  unsigned short* xvT  = xk + headElems;       // [b][h][d][n]
  unsigned short* attn = xvT + headElems;      // [4096][1024] bf16

  const int M = B_SZ * N_SEQ;  // 4096
  dim3 blk(128);

  gemm_proj<float, 0, EMB, EMB><<<dim3(EMB / 64, M / 64), blk, 0, stream>>>(
      q, Wq, bq, xq, nullptr, nullptr, nullptr);

  gemm_proj<float, 1, EMB, 2 * EMB><<<dim3(2 * EMB / 64, M / 64), blk, 0, stream>>>(
      kv, Wkv, bkv, nullptr, xk, xvT, nullptr);

  attn_kernel<<<dim3(N_SEQ / 64, B_SZ * NHEAD), blk, 0, stream>>>(
      xq, xk, xvT, mask, attn);

  gemm_proj<unsigned short, 2, EMB, EMB><<<dim3(EMB / 64, M / 64), blk, 0, stream>>>(
      attn, Wo, bo, nullptr, nullptr, nullptr, out);
}
